// ConditionalNeuralSDEGenerator_34900904247782
// MI455X (gfx1250) — matrix operand order repaired and checked by reading, not compiled, not run on hardware
//
#include <hip/hip_runtime.h>
#include <hip/hip_bf16.h>

typedef __attribute__((ext_vector_type(16))) _Float16 v16h;
typedef __attribute__((ext_vector_type(8)))  _Float16 v8h;
typedef __attribute__((ext_vector_type(8)))  float    v8f;

#define RD      256
#define BATCH   256
#define NLAGS   512
#define NSTEPS  511   // NLAGS-1
#define NPAST   128
#define BPAD    264   // f16 row stride (256 + 8) -> 528B rows, conflict-free b128 reads
#define RPAD32  260   // f32 row stride (256 + 4)
#define TMAIN   512   // main kernel block size: 16 waves, one 16-col N-tile each

static __device__ __forceinline__ v16h cat16(v8h lo, v8h hi) {
    return __builtin_shufflevector(lo, hi, 0,1,2,3,4,5,6,7,8,9,10,11,12,13,14,15);
}

// CDNA5 hardware transcendental tanh (V_TANH_F32) instead of ocml expansion
static __device__ __forceinline__ float htanh(float x) {
    return __builtin_amdgcn_tanhf(x);
}

// ---------------------------------------------------------------------------
// Kernel 1: Brownian increments  dW[t][b]  (t-major for coalesced per-step reads)
// dW[:,0] = inc[:,0]+inc[:,1];  dW[:,t] = inc[:,t+1] for t>=1
// ---------------------------------------------------------------------------
__global__ __launch_bounds__(256) void k_dw(const float* __restrict__ inc,
                                            float* __restrict__ dW) {
    int idx = blockIdx.x * 256 + threadIdx.x;
    if (idx >= NSTEPS * BATCH) return;
    int t = idx / BATCH, b = idx % BATCH;
    float v = (t == 0) ? inc[b * NLAGS + 0] + inc[b * NLAGS + 1]
                       : inc[b * NLAGS + t + 1];
    dW[t * BATCH + b] = v;
}

// ---------------------------------------------------------------------------
// Kernel 2: randomized-signature scan (sequential, single block, 256 threads)
// Z_{t+1}[i] = Z[i] + tanh(A1[i,:]Z + xi1[i]) + tanh(A2[i,:]Z + xi2[i]) * dx_t
// then rsig = tanh(Z W2h^T + b2h) W2o^T + b2o   (226 values)
// ---------------------------------------------------------------------------
__global__ __launch_bounds__(256) void k_rsig(const float* __restrict__ x_past,
                                              const float* __restrict__ A1,
                                              const float* __restrict__ A2,
                                              const float* __restrict__ xi1,
                                              const float* __restrict__ xi2,
                                              const float* __restrict__ W2h,
                                              const float* __restrict__ b2h,
                                              const float* __restrict__ W2o,
                                              const float* __restrict__ b2o,
                                              float* __restrict__ rsig) {
    __shared__ float Z[2][RD];
    __shared__ float h[32];
    const int i = threadIdx.x;
    Z[0][i] = 0.0f;
    __syncthreads();
    int cur = 0;
    const float* a1r = A1 + i * RD;
    const float* a2r = A2 + i * RD;
    const float x1 = xi1[i], x2 = xi2[i];
    for (int t = 0; t < NPAST - 1; ++t) {
        float dxt = x_past[t + 1] - x_past[t];
        float s1 = x1, s2 = x2;
        for (int j = 0; j < RD; ++j) {
            float z = Z[cur][j];
            s1 += a1r[j] * z;
            s2 += a2r[j] * z;
        }
        Z[1 - cur][i] = Z[cur][i] + htanh(s1) + htanh(s2) * dxt;
        cur ^= 1;
        __syncthreads();
    }
    if (i < 32) {
        float s = b2h[i];
        for (int j = 0; j < RD; ++j) s += Z[cur][j] * W2h[i * RD + j];
        h[i] = htanh(s);
    }
    __syncthreads();
    if (i < RD - 30) {  // 226 outputs
        float s = b2o[i];
        for (int k = 0; k < 32; ++k) s += h[k] * W2o[i * 32 + k];
        rsig[i] = s;
    }
}

// ---------------------------------------------------------------------------
// Kernel 3: assemble R0[b, :] = [ rsig(226) ; V_b(30) ],  V = MLP(V_noise_b)
// one block per batch row
// ---------------------------------------------------------------------------
__global__ __launch_bounds__(256) void k_r0(const float* __restrict__ V_noise,
                                            const float* __restrict__ W1h,
                                            const float* __restrict__ b1h,
                                            const float* __restrict__ W1o,
                                            const float* __restrict__ b1o,
                                            const float* __restrict__ rsig,
                                            float* __restrict__ R0) {
    __shared__ float h[32];
    const int b = blockIdx.x, n = threadIdx.x;
    if (n < 32) {
        float s = b1h[n];
        for (int j = 0; j < 32; ++j) s += V_noise[b * 32 + j] * W1h[n * 32 + j];
        h[n] = htanh(s);
    }
    __syncthreads();
    float val;
    if (n < RD - 30) {
        val = rsig[n];
    } else {
        int o = n - (RD - 30);
        float s = b1o[o];
        for (int k = 0; k < 32; ++k) s += h[k] * W1o[o * 32 + k];
        val = s;
    }
    R0[b * RD + n] = val;
}

// ---------------------------------------------------------------------------
// Kernel 4: main SDE recurrence + per-step readout.
// 16 blocks x 512 threads (16 waves, 4 per SIMD32 to hide LDS->WMMA latency).
// Block handles 16 batch rows; each wave owns one 16-column N-tile.
// B1,B2 live in LDS as f16; state in LDS f32 (master) + f16 (WMMA A operand).
// Per step per wave: 8 K-chunks x 2 matrices = 16x v_wmma_f32_16x16x32_f16,
// then V_TANH_F32 hardware activations on the serial update path.
// ---------------------------------------------------------------------------
__global__ __launch_bounds__(TMAIN) void k_sde(const float* __restrict__ B1,
                                               const float* __restrict__ B2,
                                               const float* __restrict__ lam1,
                                               const float* __restrict__ lam2,
                                               const float* __restrict__ Wr,
                                               const float* __restrict__ brp,
                                               const float* __restrict__ r1p,
                                               const float* __restrict__ r2p,
                                               const float* __restrict__ r3p,
                                               const float* __restrict__ r4p,
                                               const float* __restrict__ R0,
                                               const float* __restrict__ dWg,
                                               float* __restrict__ out) {
    __shared__ __align__(16) _Float16 B1s[RD * BPAD];   // 132KB, layout [n][k]
    __shared__ __align__(16) _Float16 B2s[RD * BPAD];   // 132KB
    __shared__ __align__(16) _Float16 Rh[16 * BPAD];    // f16 state (A operand)
    __shared__ float Rf[16 * RPAD32];                   // f32 master state
    __shared__ float lam1s[RD], lam2s[RD], Wrs[RD];
    __shared__ float part[32][17];
    __shared__ float dws[16];

    const int tid  = threadIdx.x;
    const int lane = tid & 31;
    const int wave = tid >> 5;          // 0..15
    const int b0   = blockIdx.x * 16;
    const float rho1 = r1p[0], rho2 = r2p[0], rho3 = r3p[0], rho4 = r4p[0];
    const float br   = brp[0];

    // preload B1/B2 (f16), lam/Wr, and the R0 tile
    for (int idx = tid; idx < RD * RD; idx += TMAIN) {
        int n = idx >> 8, k = idx & 255;
        B1s[n * BPAD + k] = (_Float16)B1[idx];
        B2s[n * BPAD + k] = (_Float16)B2[idx];
    }
    for (int idx = tid; idx < RD; idx += TMAIN) {
        lam1s[idx] = lam1[idx]; lam2s[idx] = lam2[idx]; Wrs[idx] = Wr[idx];
    }
    for (int idx = tid; idx < 16 * RD; idx += TMAIN) {
        int m = idx >> 8, n = idx & 255;
        float v = R0[(b0 + m) * RD + n];
        Rf[m * RPAD32 + n] = v;
        Rh[m * BPAD + n]   = (_Float16)v;
    }
    __syncthreads();

    // readout at t = 0  (32 segments of 8 columns each)
    {
        int m = tid & 15, seg = tid >> 4;
        float p = 0.0f;
        for (int j = 0; j < 8; ++j)
            p += Rf[m * RPAD32 + seg * 8 + j] * Wrs[seg * 8 + j];
        part[seg][m] = p;
        __syncthreads();
        if (tid < 16) {
            float s = br;
            for (int sg = 0; sg < 32; ++sg) s += part[sg][tid];
            out[(b0 + tid) * NLAGS + 0] = s;
        }
        __syncthreads();
    }

    const int hi = lane >> 4;   // lane half: selects contiguous 16-wide K sub-block (k = 16*hi + s)
    const int li = lane & 15;   // row (A) / column (B,D) within tile
    const int n0 = wave * 16;   // this wave's 16-column N tile

    for (int t = 1; t < NLAGS; ++t) {
        if (tid < 16) dws[tid] = dWg[(t - 1) * BATCH + b0 + tid];

        v8f acc1 = {0,0,0,0,0,0,0,0};   // B1 contribution
        v8f acc2 = {0,0,0,0,0,0,0,0};   // B2 contribution

        for (int kc = 0; kc < 8; ++kc) {
            // A fragment: lane<16 -> K = 32kc+{0..15}; lane>=16 -> 32kc+{16..31}
            const int kA = kc * 32 + hi * 16;
            v8h alo = *(const v8h*)&Rh[li * BPAD + kA];
            v8h ahi = *(const v8h*)&Rh[li * BPAD + kA + 8];
            v16h a  = cat16(alo, ahi);

            // B fragment: B = Bx^T, so lane needs 16 contiguous K of Bx row n
            const int kB = kc * 32 + hi * 16;
            const _Float16* p1 = &B1s[(n0 + li) * BPAD + kB];
            const _Float16* p2 = &B2s[(n0 + li) * BPAD + kB];
            v16h b1 = cat16(*(const v8h*)p1, *(const v8h*)(p1 + 8));
            v16h b2 = cat16(*(const v8h*)p2, *(const v8h*)(p2 + 8));
            acc1 = __builtin_amdgcn_wmma_f32_16x16x32_f16(false, a, false, b1,
                                                          (short)0, acc1, false, false);
            acc2 = __builtin_amdgcn_wmma_f32_16x16x32_f16(false, a, false, b2,
                                                          (short)0, acc2, false, false);
        }
        __syncthreads();   // all Rh reads done before overwrite; dws visible

        // update this wave's 16 columns (D layout: VGPR r -> M = r + 8*hi, lane -> N)
        for (int r = 0; r < 8; ++r) {
            int m = r + 8 * hi;
            int n = n0 + li;
            float rn = Rf[m * RPAD32 + n]
                     + htanh(rho1 * acc1[r] + rho2 * lam1s[n])
                     + htanh(rho3 * acc2[r] + rho4 * lam2s[n]) * dws[m];
            Rf[m * RPAD32 + n] = rn;
            Rh[m * BPAD + n]   = (_Float16)rn;
        }
        __syncthreads();

        // readout x[b, t] = R_t[b,:] . Wr + br
        {
            int m = tid & 15, seg = tid >> 4;
            float p = 0.0f;
            for (int j = 0; j < 8; ++j)
                p += Rf[m * RPAD32 + seg * 8 + j] * Wrs[seg * 8 + j];
            part[seg][m] = p;
        }
        __syncthreads();
        if (tid < 16) {
            float s = br;
            for (int sg = 0; sg < 32; ++sg) s += part[sg][tid];
            out[(b0 + tid) * NLAGS + t] = s;
        }
        __syncthreads();
    }
}

// ---------------------------------------------------------------------------
extern "C" void kernel_launch(void* const* d_in, const int* in_sizes, int n_in,
                              void* d_out, int out_size, void* d_ws, size_t ws_size,
                              hipStream_t stream) {
    const float* x_past  = (const float*)d_in[2];
    const float* V_noise = (const float*)d_in[3];
    const float* inc     = (const float*)d_in[4];
    const float* W1h     = (const float*)d_in[5];
    const float* b1h     = (const float*)d_in[6];
    const float* W1o     = (const float*)d_in[7];
    const float* b1o     = (const float*)d_in[8];
    const float* W2h     = (const float*)d_in[9];
    const float* b2h     = (const float*)d_in[10];
    const float* W2o     = (const float*)d_in[11];
    const float* b2o     = (const float*)d_in[12];
    const float* rho1    = (const float*)d_in[13];
    const float* rho2    = (const float*)d_in[14];
    const float* rho3    = (const float*)d_in[15];
    const float* rho4    = (const float*)d_in[16];
    const float* B1      = (const float*)d_in[17];
    const float* B2      = (const float*)d_in[18];
    const float* lam1    = (const float*)d_in[19];
    const float* lam2    = (const float*)d_in[20];
    const float* A1      = (const float*)d_in[21];
    const float* A2      = (const float*)d_in[22];
    const float* xi1     = (const float*)d_in[23];
    const float* xi2     = (const float*)d_in[24];
    const float* Wr      = (const float*)d_in[25];
    const float* br      = (const float*)d_in[26];

    float* ws   = (float*)d_ws;
    float* dW   = ws;                               // 511*256
    float* R0   = dW + NSTEPS * BATCH;              // 256*256
    float* rsig = R0 + BATCH * RD;                  // 226

    k_dw  <<<(NSTEPS * BATCH + 255) / 256, 256, 0, stream>>>(inc, dW);
    k_rsig<<<1,   256, 0, stream>>>(x_past, A1, A2, xi1, xi2, W2h, b2h, W2o, b2o, rsig);
    k_r0  <<<256, 256, 0, stream>>>(V_noise, W1h, b1h, W1o, b1o, rsig, R0);
    k_sde <<<16,  TMAIN, 0, stream>>>(B1, B2, lam1, lam2, Wr, br,
                                      rho1, rho2, rho3, rho4, R0, dW, (float*)d_out);
}
